// FasterRCNN_21869973471798
// MI455X (gfx1250) — matrix operand order repaired and checked by reading, not compiled, not run on hardware
//
#include <hip/hip_runtime.h>
#include <hip/hip_bf16.h>
#include <math.h>

// ---------------- types for WMMA ----------------
typedef __attribute__((ext_vector_type(16))) __bf16 v16bf;
typedef __attribute__((ext_vector_type(8)))  __bf16 v8bf;
typedef __attribute__((ext_vector_type(8)))  float  v8f;

#define PNUM   2500   // 50*50 positions
#define HW     50
#define CFEAT  256
#define TOPK_N 300
#define DFLAT  12544  // 256*7*7
#define FCDIM  4096
#define NCLS   21     // CLASS_NUM + 1
#define NOUT   25     // 21 + 4
#define LDS_STRIDE 40 // bf16 elems per row (80B) -> conflict-free b128 reads

#define WMMA_BF16(A_, B_, C_) \
  __builtin_amdgcn_wmma_f32_16x16x32_bf16(false, (A_), false, (B_), (short)0, (C_), false, false)

__device__ __forceinline__ __bf16 f2bf(float f) { return (__bf16)f; }

// K offset of A-fragment vgpr-pair j (j=0..7), contiguous order k = 16*half + 2*j
__device__ __forceinline__ int a_kbase(int j, int half) {
  return 2 * j + 16 * half;
}

__device__ __forceinline__ v16bf cvt_b_frag(const float4 b[4]) {
  v16bf r;
#pragma unroll
  for (int q = 0; q < 4; ++q) {
    r[4 * q + 0] = f2bf(b[q].x);
    r[4 * q + 1] = f2bf(b[q].y);
    r[4 * q + 2] = f2bf(b[q].z);
    r[4 * q + 3] = f2bf(b[q].w);
  }
  return r;
}

__device__ __forceinline__ v16bf cvt_a_pairs(const float2 a[8], float msk) {
  v16bf r;
#pragma unroll
  for (int j = 0; j < 8; ++j) {
    r[2 * j + 0] = f2bf(a[j].x * msk);
    r[2 * j + 1] = f2bf(a[j].y * msk);
  }
  return r;
}

__device__ __forceinline__ v8bf cvt_stage(float4 f0, float4 f1, float msk) {
  v8bf v;
  v[0] = f2bf(f0.x * msk); v[1] = f2bf(f0.y * msk);
  v[2] = f2bf(f0.z * msk); v[3] = f2bf(f0.w * msk);
  v[4] = f2bf(f1.x * msk); v[5] = f2bf(f1.y * msk);
  v[6] = f2bf(f1.z * msk); v[7] = f2bf(f1.w * msk);
  return v;
}

// ================= FC GEMM: LDS-staged A, 8 waves share one 64-row M-block =========
// Block: 256 thr. Wave w -> N-tile (nb8*8 + w). A slice (64 x 32) staged in LDS as bf16.
// 2-deep pipeline: raw A slice for LDS-store is loaded one full iteration ahead.
__global__ void fc_gemm_kernel(const float* __restrict__ A,
                               const float* __restrict__ Wt,
                               const float* __restrict__ bias,
                               float* __restrict__ out,
                               int M, int K, int N, int Mblocks, int relu) {
  __shared__ __bf16 lds_a[2][64 * LDS_STRIDE];
  int tid = threadIdx.x;
  int wave = tid >> 5, lane = tid & 31;
  int mb = blockIdx.x % Mblocks;
  int nb8 = blockIdx.x / Mblocks;
  int m0 = mb * 64;
  int half = lane >> 4, l15 = lane & 15;
  int ncol = (nb8 * 8 + wave) * 16 + l15;
  const float* wrow = Wt + (size_t)ncol * K;

  // staging role: row = tid/4 (0..63), seg = tid%4 (8 consecutive K)
  int srow = tid >> 2, sseg = tid & 3;
  int grow = m0 + srow;
  float smsk = (grow < M) ? 1.0f : 0.0f;
  const float* arow = A + (size_t)min(grow, M - 1) * K + sseg * 8;
  __bf16* sdst = &lds_a[0][srow * LDS_STRIDE + sseg * 8];
  const int sbuf_off = 64 * LDS_STRIDE;

  v8f c0 = {}, c1 = {}, c2 = {}, c3 = {};
  float4 bcur[4];
  {
    const float4* bp = (const float4*)(wrow + 16 * half);
#pragma unroll
    for (int i = 0; i < 4; ++i) bcur[i] = bp[i];
  }
  // stage k0=0 into buffer 0; pre-load raw slice for k=32
  {
    float4 f0 = *(const float4*)(arow);
    float4 f1 = *(const float4*)(arow + 4);
    *(v8bf*)sdst = cvt_stage(f0, f1, smsk);
  }
  float4 sf0, sf1;   // raw slice destined for the *next* LDS store
  {
    int k1 = (32 < K) ? 32 : 0;
    sf0 = *(const float4*)(arow + k1);
    sf1 = *(const float4*)(arow + k1 + 4);
  }
  __syncthreads();

  int buf = 0;
#pragma clang loop unroll(disable)
  for (int k0 = 0; k0 < K; k0 += 32) {
    int kn = (k0 + 32 < K) ? (k0 + 32) : 0;     // next slice (for B regs + LDS store)
    int kn2 = (k0 + 64 < K) ? (k0 + 64) : 0;    // slice after next (raw A preload)
    // issue all next-iteration global loads up front
    float4 nf0 = *(const float4*)(arow + kn2);
    float4 nf1 = *(const float4*)(arow + kn2 + 4);
    float4 bnxt[4];
    {
      const float4* bp = (const float4*)(wrow + kn + 16 * half);
#pragma unroll
      for (int i = 0; i < 4; ++i) bnxt[i] = bp[i];
    }
    __builtin_prefetch((const void*)(wrow + k0 + 544), 0, 0);

    // compute: fetch ALL A fragments from LDS first, then 4 back-to-back WMMAs
    const __bf16* lbase = &lds_a[0][buf ? sbuf_off : 0];
    v8bf lo[4], hi[4];
#pragma unroll
    for (int s = 0; s < 4; ++s) {
      int row = s * 16 + l15;
      lo[s] = *(const v8bf*)(lbase + row * LDS_STRIDE + 16 * half);
      hi[s] = *(const v8bf*)(lbase + row * LDS_STRIDE + 16 * half + 8);
    }
    v16bf b = cvt_b_frag(bcur);
    c0 = WMMA_BF16(__builtin_shufflevector(lo[0], hi[0], 0, 1, 2, 3, 4, 5, 6, 7,
                                           8, 9, 10, 11, 12, 13, 14, 15), b, c0);
    c1 = WMMA_BF16(__builtin_shufflevector(lo[1], hi[1], 0, 1, 2, 3, 4, 5, 6, 7,
                                           8, 9, 10, 11, 12, 13, 14, 15), b, c1);
    c2 = WMMA_BF16(__builtin_shufflevector(lo[2], hi[2], 0, 1, 2, 3, 4, 5, 6, 7,
                                           8, 9, 10, 11, 12, 13, 14, 15), b, c2);
    c3 = WMMA_BF16(__builtin_shufflevector(lo[3], hi[3], 0, 1, 2, 3, 4, 5, 6, 7,
                                           8, 9, 10, 11, 12, 13, 14, 15), b, c3);

    // stage slice kn (raw loaded LAST iteration -> no wait here)
    *(v8bf*)(sdst + (buf ? 0 : sbuf_off)) = cvt_stage(sf0, sf1, smsk);
    __syncthreads();
    buf ^= 1;
    sf0 = nf0; sf1 = nf1;
#pragma unroll
    for (int i = 0; i < 4; ++i) bcur[i] = bnxt[i];
  }

  float bv = bias[ncol];
  v8f accs[4] = {c0, c1, c2, c3};
#pragma unroll
  for (int s = 0; s < 4; ++s) {
#pragma unroll
    for (int j = 0; j < 8; ++j) {
      int m = m0 + s * 16 + j + 8 * half;
      if (m < M) {
        float v = accs[s][j] + bv;
        if (relu) v = fmaxf(v, 0.0f);
        out[(size_t)m * N + ncol] = v;
      }
    }
  }
}

// ================= Conv1: stride-16 patchify (1 M-tile x 4 N-tiles) =================
__device__ __forceinline__ void c1_load_a(float2 a[8], const float* __restrict__ x,
                                          int oy, int ox, int k0, int half) {
#pragma unroll
  for (int j = 0; j < 8; ++j) {
    int ka = k0 + a_kbase(j, half);
    int ci = ka >> 8, rem = ka & 255;
    int py = rem >> 4, px = rem & 15;           // px even -> pair contiguous
    a[j] = *(const float2*)(x + ci * 640000 + (oy * 16 + py) * 800 + ox * 16 + px);
  }
}

__global__ void conv_feat_kernel(const float* __restrict__ x,
                                 const float* __restrict__ Wf,
                                 const float* __restrict__ bf,
                                 float* __restrict__ feature) {
  const int Mtiles = 157, K = 768;
  int wave = threadIdx.x >> 5, lane = threadIdx.x & 31;
  int t = blockIdx.x * (blockDim.x >> 5) + wave;
  if (t >= Mtiles * 4) return;
  int mt = t % Mtiles, nb = t / Mtiles;
  int m0 = mt * 16;
  int half = lane >> 4, l15 = lane & 15;
  int mrow = m0 + l15;
  float msk = (mrow < PNUM) ? 1.0f : 0.0f;
  int mrc = min(mrow, PNUM - 1);
  int oy = mrc / HW, ox = mrc % HW;
  const float* w[4];
#pragma unroll
  for (int q = 0; q < 4; ++q) w[q] = Wf + (size_t)(nb * 64 + q * 16 + l15) * K;
  v8f c0 = {}, c1 = {}, c2 = {}, c3 = {};
  float2 acur[8];
  c1_load_a(acur, x, oy, ox, 0, half);
#pragma clang loop unroll(disable)
  for (int k0 = 0; k0 < K; k0 += 32) {
    int kn = (k0 + 32 < K) ? (k0 + 32) : 0;
    float2 anxt[8];
    c1_load_a(anxt, x, oy, ox, kn, half);
    float4 braw[4][4];
#pragma unroll
    for (int q = 0; q < 4; ++q) {
      const float4* bp = (const float4*)(w[q] + k0 + 16 * half);
#pragma unroll
      for (int i = 0; i < 4; ++i) braw[q][i] = bp[i];
    }
    v16bf a = cvt_a_pairs(acur, msk);
    c0 = WMMA_BF16(a, cvt_b_frag(braw[0]), c0);
    c1 = WMMA_BF16(a, cvt_b_frag(braw[1]), c1);
    c2 = WMMA_BF16(a, cvt_b_frag(braw[2]), c2);
    c3 = WMMA_BF16(a, cvt_b_frag(braw[3]), c3);
#pragma unroll
    for (int j = 0; j < 8; ++j) acur[j] = anxt[j];
  }
  v8f accs[4] = {c0, c1, c2, c3};
#pragma unroll
  for (int q = 0; q < 4; ++q) {
    int nc = nb * 64 + q * 16 + l15;
    float bias = bf[nc];
#pragma unroll
    for (int j = 0; j < 8; ++j) {
      int m = m0 + j + 8 * half;
      if (m < PNUM) feature[nc * PNUM + m] = accs[q][j] + bias;
    }
  }
}

// ================= Conv2: 3x3 SAME + ReLU (1 M-tile x 4 N-tiles) ====================
__device__ __forceinline__ unsigned c2_load_a(float a[16], const float* __restrict__ feature,
                                              int y, int xq, int k0, int half, float rowmsk) {
  unsigned mbits = 0;
#pragma unroll
  for (int e = 0; e < 16; ++e) {
    int j = e >> 1;
    int ka = k0 + a_kbase(j, half) + (e & 1);
    int ci = ka / 9;
    int rr = ka - ci * 9;
    int dy = rr / 3 - 1;
    int dx = rr - (rr / 3) * 3 - 1;
    int yy = y + dy, xx = xq + dx;
    bool inb = ((unsigned)yy < (unsigned)HW) && ((unsigned)xx < (unsigned)HW) && (rowmsk != 0.0f);
    mbits |= (inb ? 1u : 0u) << e;
    int yyc = min(max(yy, 0), HW - 1);
    int xxc = min(max(xx, 0), HW - 1);
    a[e] = feature[ci * PNUM + yyc * HW + xxc];
  }
  return mbits;
}

__global__ void conv_hmap_kernel(const float* __restrict__ feature,
                                 const float* __restrict__ W1,
                                 const float* __restrict__ b1,
                                 float* __restrict__ hmap) {
  const int Mtiles = 157, K = 2304;
  int wave = threadIdx.x >> 5, lane = threadIdx.x & 31;
  int t = blockIdx.x * (blockDim.x >> 5) + wave;
  if (t >= Mtiles * 4) return;
  int mt = t % Mtiles, nb = t / Mtiles;
  int m0 = mt * 16;
  int half = lane >> 4, l15 = lane & 15;
  int mrow = m0 + l15;
  float msk = (mrow < PNUM) ? 1.0f : 0.0f;
  int mrc = min(mrow, PNUM - 1);
  int y = mrc / HW, xq = mrc % HW;
  const float* w[4];
#pragma unroll
  for (int q = 0; q < 4; ++q) w[q] = W1 + (size_t)(nb * 64 + q * 16 + l15) * K;
  v8f c0 = {}, c1 = {}, c2 = {}, c3 = {};
  float acur[16];
  unsigned mcur = c2_load_a(acur, feature, y, xq, 0, half, msk);
#pragma clang loop unroll(disable)
  for (int k0 = 0; k0 < K; k0 += 32) {
    int kn = (k0 + 32 < K) ? (k0 + 32) : 0;
    float anxt[16];
    unsigned mnxt = c2_load_a(anxt, feature, y, xq, kn, half, msk);
    float4 braw[4][4];
#pragma unroll
    for (int q = 0; q < 4; ++q) {
      const float4* bp = (const float4*)(w[q] + k0 + 16 * half);
#pragma unroll
      for (int i = 0; i < 4; ++i) braw[q][i] = bp[i];
    }
    v16bf a;
#pragma unroll
    for (int e = 0; e < 16; ++e)
      a[e] = f2bf(((mcur >> e) & 1u) ? acur[e] : 0.0f);
    c0 = WMMA_BF16(a, cvt_b_frag(braw[0]), c0);
    c1 = WMMA_BF16(a, cvt_b_frag(braw[1]), c1);
    c2 = WMMA_BF16(a, cvt_b_frag(braw[2]), c2);
    c3 = WMMA_BF16(a, cvt_b_frag(braw[3]), c3);
#pragma unroll
    for (int e = 0; e < 16; ++e) acur[e] = anxt[e];
    mcur = mnxt;
  }
  v8f accs[4] = {c0, c1, c2, c3};
#pragma unroll
  for (int q = 0; q < 4; ++q) {
    int nc = nb * 64 + q * 16 + l15;
    float bias = b1[nc];
#pragma unroll
    for (int j = 0; j < 8; ++j) {
      int m = m0 + j + 8 * half;
      if (m < PNUM) hmap[m * CFEAT + nc] = fmaxf(accs[q][j] + bias, 0.0f);
    }
  }
}

// ---------------- Kernel 3: cls/box heads + softmax score ----------------
__global__ void head_kernel(const float* __restrict__ hmap,
                            const float* __restrict__ Wc, const float* __restrict__ bc,
                            const float* __restrict__ Wb, const float* __restrict__ bb,
                            float* __restrict__ scores, float* __restrict__ preb) {
  int p = blockIdx.x * blockDim.x + threadIdx.x;
  if (p >= PNUM) return;
  const float* h = hmap + p * CFEAT;
  float l0 = 0.f, l1 = 0.f, v0 = 0.f, v1 = 0.f, v2 = 0.f, v3 = 0.f;
#pragma unroll 4
  for (int c = 0; c < CFEAT; ++c) {
    float hv = h[c];
    l0 += hv * Wc[c];
    l1 += hv * Wc[CFEAT + c];
    v0 += hv * Wb[c];
    v1 += hv * Wb[CFEAT + c];
    v2 += hv * Wb[2 * CFEAT + c];
    v3 += hv * Wb[3 * CFEAT + c];
  }
  l0 += bc[0]; l1 += bc[1];
  scores[p] = 1.0f / (1.0f + expf(l0 - l1));
  preb[p * 4 + 0] = v0 + bb[0];
  preb[p * 4 + 1] = v1 + bb[1];
  preb[p * 4 + 2] = v2 + bb[2];
  preb[p * 4 + 3] = v3 + bb[3];
}

// ---------------- Kernel 4: top-300 via rank counting ----------------
__global__ void topk_kernel(const float* __restrict__ s,
                            float* __restrict__ ts, int* __restrict__ ti) {
  int i = blockIdx.x * blockDim.x + threadIdx.x;
  if (i >= PNUM) return;
  float si = s[i];
  int rank = 0;
  for (int j = 0; j < PNUM; ++j) {
    float sj = s[j];
    rank += (sj > si) || (sj == si && j < i);
  }
  if (rank < TOPK_N) { ts[rank] = si; ti[rank] = i; }
}

// ---------------- Kernel 5: box decode + sequential NMS (single block) -------------
__global__ void nms_kernel(const float* __restrict__ preb,
                           const float* __restrict__ ts, const int* __restrict__ ti,
                           const int* __restrict__ wp, const int* __restrict__ hp,
                           float* __restrict__ boxes, int* __restrict__ keep) {
  __shared__ float sb[TOPK_N * 4];
  __shared__ float sarea[TOPK_N];
  __shared__ int skeep[TOPK_N];
  __shared__ int svalid[TOPK_N];
  int r = threadIdx.x;
  if (r < TOPK_N) {
    int id = ti[r];
    float p0 = preb[id * 4 + 0], p1 = preb[id * 4 + 1];
    float p2 = preb[id * 4 + 2], p3 = preb[id * 4 + 3];
    float wf = (float)wp[0], hf = (float)hp[0];
    float x1 = fmaxf(p2 - p0 * 0.5f, 0.0f);
    float y1 = fmaxf(p3 - p1 * 0.5f, 0.0f);
    float x2 = fminf(p2 + p0, wf);
    float y2 = fminf(p3 + p1, hf);
    sb[r * 4 + 0] = x1; sb[r * 4 + 1] = y1; sb[r * 4 + 2] = x2; sb[r * 4 + 3] = y2;
    sarea[r] = fmaxf(x2 - x1, 0.0f) * fmaxf(y2 - y1, 0.0f);
    svalid[r] = (ts[r] > 0.5f) ? 1 : 0;
    skeep[r] = svalid[r];
    boxes[r * 4 + 0] = x1; boxes[r * 4 + 1] = y1;
    boxes[r * 4 + 2] = x2; boxes[r * 4 + 3] = y2;
  }
  __syncthreads();
  for (int i = 0; i < TOPK_N; ++i) {
    int ki = skeep[i] & svalid[i];
    if (r < TOPK_N && r > i && ki) {
      float ix1 = fmaxf(sb[i * 4 + 0], sb[r * 4 + 0]);
      float iy1 = fmaxf(sb[i * 4 + 1], sb[r * 4 + 1]);
      float ix2 = fminf(sb[i * 4 + 2], sb[r * 4 + 2]);
      float iy2 = fminf(sb[i * 4 + 3], sb[r * 4 + 3]);
      float inter = fmaxf(ix2 - ix1, 0.0f) * fmaxf(iy2 - iy1, 0.0f);
      float iou = inter / fmaxf(sarea[i] + sarea[r] - inter, 1e-6f);
      if (iou > 0.7f) skeep[r] = 0;
    }
    __syncthreads();
  }
  if (r < TOPK_N) keep[r] = skeep[r];
}

// ---------------- Kernel 6: ROI max pool (2x2 samples) ----------------
__global__ void roi_kernel(const float* __restrict__ feature,
                           const float* __restrict__ boxes,
                           float* __restrict__ flat) {
  __shared__ float sbox[4];
  __shared__ int six[14], siy[14];
  int rblk = blockIdx.x;     // 0..299
  int c = threadIdx.x;       // 0..255
  if (c < 4) sbox[c] = boxes[rblk * 4 + c] * (1.0f / 16.0f);
  __syncthreads();
  if (c < 14) {
    int cell = c >> 1, samp = c & 1;
    float frac = ((float)cell + ((float)samp + 0.5f) * 0.5f) / 7.0f;
    float xs = sbox[0] + frac * (sbox[2] - sbox[0]);
    float ys = sbox[1] + frac * (sbox[3] - sbox[1]);
    six[c] = min(max((int)floorf(xs), 0), HW - 1);
    siy[c] = min(max((int)floorf(ys), 0), HW - 1);
  }
  __syncthreads();
  const float* f = feature + c * PNUM;
  float* o = flat + rblk * DFLAT + c * 49;
  for (int oy = 0; oy < 7; ++oy) {
    for (int ox = 0; ox < 7; ++ox) {
      float m = -INFINITY;
#pragma unroll
      for (int sy = 0; sy < 2; ++sy)
#pragma unroll
        for (int sx = 0; sx < 2; ++sx)
          m = fmaxf(m, f[siy[oy * 2 + sy] * HW + six[ox * 2 + sx]]);
      o[oy * 7 + ox] = m;
    }
  }
}

// ---------------- Kernel 8: final cls/box heads + keep mask ----------------
__global__ void final_kernel(const float* __restrict__ h2,
                             const float* __restrict__ cls_w, const float* __restrict__ cls_b,
                             const float* __restrict__ box_w, const float* __restrict__ box_b,
                             const int* __restrict__ keep,
                             float* __restrict__ out) {
  int idx = blockIdx.x * blockDim.x + threadIdx.x;
  if (idx >= TOPK_N * NOUT) return;
  int r = idx / NOUT, o = idx % NOUT;
  const float* w;
  float b;
  if (o < NCLS) { w = cls_w + (size_t)o * FCDIM; b = cls_b[o]; }
  else          { w = box_w + (size_t)(o - NCLS) * FCDIM; b = box_b[o - NCLS]; }
  const float* h = h2 + (size_t)r * FCDIM;
  float acc = 0.0f;
#pragma unroll 4
  for (int k = 0; k < FCDIM; ++k) acc += h[k] * w[k];
  acc += b;
  out[idx] = keep[r] ? acc : 0.0f;
}

// ---------------- launch ----------------
extern "C" void kernel_launch(void* const* d_in, const int* in_sizes, int n_in,
                              void* d_out, int out_size, void* d_ws, size_t ws_size,
                              hipStream_t stream) {
  const float* x      = (const float*)d_in[0];
  // d_in[1] = gt_boxes (unused by reference)
  const float* W_feat = (const float*)d_in[2];
  const float* b_feat = (const float*)d_in[3];
  const float* W1     = (const float*)d_in[4];
  const float* b1     = (const float*)d_in[5];
  const float* Wc     = (const float*)d_in[6];
  const float* bc     = (const float*)d_in[7];
  const float* Wb     = (const float*)d_in[8];
  const float* bb     = (const float*)d_in[9];
  const float* fc1_w  = (const float*)d_in[10];
  const float* fc1_b  = (const float*)d_in[11];
  const float* fc2_w  = (const float*)d_in[12];
  const float* fc2_b  = (const float*)d_in[13];
  const float* cls_w  = (const float*)d_in[14];
  const float* cls_b  = (const float*)d_in[15];
  const float* box_w  = (const float*)d_in[16];
  const float* box_b  = (const float*)d_in[17];
  const int*   wp     = (const int*)d_in[18];
  const int*   hp     = (const int*)d_in[19];

  char* ws = (char*)d_ws;
  size_t off = 0;
  auto alloc = [&](size_t bytes) -> char* {
    char* p = ws + off;
    off = (off + bytes + 255) & ~(size_t)255;
    return p;
  };
  float* feature = (float*)alloc((size_t)CFEAT * PNUM * 4);   // [c][p]
  float* hmap    = (float*)alloc((size_t)PNUM * CFEAT * 4);   // [p][c]
  float* scores  = (float*)alloc((size_t)PNUM * 4);
  float* preb    = (float*)alloc((size_t)PNUM * 4 * 4);
  float* tscore  = (float*)alloc((size_t)TOPK_N * 4);
  int*   tidx    = (int*)  alloc((size_t)TOPK_N * 4);
  float* boxes   = (float*)alloc((size_t)TOPK_N * 4 * 4);
  int*   keep    = (int*)  alloc((size_t)TOPK_N * 4);
  float* flat    = (float*)alloc((size_t)TOPK_N * DFLAT * 4);
  float* h1      = (float*)alloc((size_t)TOPK_N * FCDIM * 4);
  float* h2      = (float*)alloc((size_t)TOPK_N * FCDIM * 4);

  // conv GEMMs: 157 M-tiles x 4 N-blocks = 628 waves -> 79 blocks of 8 waves
  conv_feat_kernel<<<79, 256, 0, stream>>>(x, W_feat, b_feat, feature);
  conv_hmap_kernel<<<79, 256, 0, stream>>>(feature, W1, b1, hmap);
  head_kernel<<<(PNUM + 255) / 256, 256, 0, stream>>>(hmap, Wc, bc, Wb, bb, scores, preb);
  topk_kernel<<<(PNUM + 255) / 256, 256, 0, stream>>>(scores, tscore, tidx);
  nms_kernel<<<1, 512, 0, stream>>>(preb, tscore, tidx, wp, hp, boxes, keep);
  roi_kernel<<<TOPK_N, 256, 0, stream>>>(feature, boxes, flat);
  // fc: grid = Mblocks(5) * N/128(32) = 160 blocks of 8 waves
  fc_gemm_kernel<<<160, 256, 0, stream>>>(flat, fc1_w, fc1_b, h1, TOPK_N, DFLAT, FCDIM, 5, 1);
  fc_gemm_kernel<<<160, 256, 0, stream>>>(h1, fc2_w, fc2_b, h2, TOPK_N, FCDIM, FCDIM, 5, 1);
  final_kernel<<<(TOPK_N * NOUT + 255) / 256, 256, 0, stream>>>(
      h2, cls_w, cls_b, box_w, box_b, keep, (float*)d_out);
}
